// MultiHeadAttention_56650618634963
// MI455X (gfx1250) — matrix operand order repaired and checked by reading, not compiled, not run on hardware
//
#include <hip/hip_runtime.h>
#include <math.h>

#define SDIM 2048
#define BDIM 2
#define EDIM 1024
#define HDIM 16
#define HD   64

typedef __attribute__((ext_vector_type(16))) __bf16 v16bf;
typedef __attribute__((ext_vector_type(8)))  float  v8f;

union FragU { v16bf v; uint4 q[2]; unsigned u[8]; };

__device__ __forceinline__ unsigned short f2bf(float f) {
  union { float f; unsigned u; } x; x.f = f;
  unsigned r = x.u + 0x7FFFu + ((x.u >> 16) & 1u);   // round-to-nearest-even
  return (unsigned short)(r >> 16);
}

// A-matrix 16x32 (MxK) fragment: lane = row (lane&15); half 0 holds K {kb..kb+7, kb+16..kb+23},
// half 1 holds K {kb+8..kb+15, kb+24..kb+31}.  Two 16B loads.
__device__ __forceinline__ v16bf load_a_frag(const unsigned short* base, int stride,
                                             int row, int kb, int lane) {
  const unsigned short* p = base + (long)(row + (lane & 15)) * stride + kb + ((lane >> 4) << 3);
  FragU f;
  f.q[0] = *(const uint4*)(p);
  f.q[1] = *(const uint4*)(p + 16);
  return f.v;
}

// B-matrix 32x16 (KxN) fragment, sourced from row-major [N,K] storage (i.e. B^T rows):
// lane = column N (lane&15); half 0 holds K {kb..kb+7, kb+16..kb+23}, half 1 K {kb+8..kb+15, kb+24..kb+31}.
__device__ __forceinline__ v16bf load_b_frag(const unsigned short* base, int stride,
                                             int row, int kb, int lane) {
  const unsigned short* p = base + (long)(row + (lane & 15)) * stride + kb + ((lane >> 4) << 3);
  FragU f;
  f.q[0] = *(const uint4*)(p);
  f.q[1] = *(const uint4*)(p + 16);
  return f.v;
}

__global__ void cvt_f32_bf16(const float* __restrict__ in, unsigned short* __restrict__ out, int n) {
  int i = blockIdx.x * blockDim.x + threadIdx.x;
  if (i < n) out[i] = f2bf(in[i]);
}

// C[m,n] = alpha * (sum_k A[m,k]*W[n,k] + bias[n]).
// Each wave: 32 rows x 64 cols (2 row groups reuse every B fragment twice).
// mode 0: fp32 out, row-major [M,EDIM] -> of32
// mode 1: bf16 out scattered to [B,H,S,HD]   (for Q, K)
// mode 2: bf16 out scattered to [B,H,HD,S]   (for V, transposed)
__global__ __launch_bounds__(128) void gemm_wmma(
    const unsigned short* __restrict__ A, const unsigned short* __restrict__ W,
    const float* __restrict__ bias, float alpha, int K, int mode,
    unsigned short* __restrict__ obf, float* __restrict__ of32)
{
  int lane = threadIdx.x & 31;
  int wave = threadIdx.x >> 5;
  int r0 = blockIdx.y * 128 + wave * 32;
  int c0 = blockIdx.x * 64;

  v8f acc[2][4];
  v8f z = {};
#pragma unroll
  for (int g = 0; g < 2; ++g)
#pragma unroll
    for (int j = 0; j < 4; ++j) acc[g][j] = z;

  for (int kb = 0; kb < K; kb += 32) {
    v16bf a0 = load_a_frag(A, K, r0,      kb, lane);
    v16bf a1 = load_a_frag(A, K, r0 + 16, kb, lane);
#pragma unroll
    for (int j = 0; j < 4; ++j) {
      v16bf bw = load_b_frag(W, K, c0 + j * 16, kb, lane);
      acc[0][j] = __builtin_amdgcn_wmma_f32_16x16x32_bf16(false, a0, false, bw,
                                                          (short)0, acc[0][j], false, false);
      acc[1][j] = __builtin_amdgcn_wmma_f32_16x16x32_bf16(false, a1, false, bw,
                                                          (short)0, acc[1][j], false, false);
    }
  }

  int half = lane >> 4;
  int ln = lane & 15;
#pragma unroll
  for (int g = 0; g < 2; ++g) {
#pragma unroll
    for (int j = 0; j < 4; ++j) {
      int n = c0 + j * 16 + ln;
      float bv = bias[n];
#pragma unroll
      for (int v = 0; v < 8; ++v) {
        int m = r0 + g * 16 + v + 8 * half;
        float o = alpha * (acc[g][j][v] + bv);
        if (mode == 0) {
          of32[(long)m * EDIM + n] = o;
        } else {
          int s = m / BDIM, b = m % BDIM;
          int h = n >> 6, hd = n & 63;
          long bh = (long)b * HDIM + h;
          long idx = (mode == 1) ? ((bh * SDIM + s) * HD + hd)
                                 : ((bh * HD + hd) * SDIM + s);
          obf[idx] = f2bf(o);
        }
      }
    }
  }
}

// Flash-style attention per (b,h).  Each wave owns 32 query rows (2 groups of 16) and
// streams keys in t-tiles of 32.  Scores are computed TRANSPOSED (S^T = K * Q^T) so that
// each lane owns one query row: softmax reductions are per-lane + one shfl_xor(16), and
// the S^T C-layout coincides with the A-fragment layout of P (pure per-lane bf16 pack,
// no LDS round trip).
// Qp, Kp: [B*H, S, HD] bf16; Vt: [B*H, HD, S] bf16; ctx: [S, B, E] bf16.
__global__ __launch_bounds__(128) void attn_wmma(
    const unsigned short* __restrict__ Qp,
    const unsigned short* __restrict__ Kp,
    const unsigned short* __restrict__ Vt,
    unsigned short* __restrict__ ctx)
{
  int lane = threadIdx.x & 31;
  int wave = threadIdx.x >> 5;
  int half = lane >> 4;
  int ln = lane & 15;
  int bh = blockIdx.y;
  int b = bh / HDIM;
  int h = bh % HDIM;
  int s0 = blockIdx.x * 128 + wave * 32;

  const unsigned short* Qb = Qp + (long)bh * SDIM * HD;
  const unsigned short* Kb = Kp + (long)bh * SDIM * HD;
  const unsigned short* Vb = Vt + (long)bh * HD * SDIM;

  // Q as B-matrix of S^T = K * Q^T: one fragment per (row group, d-step), loaded once.
  v16bf qB[2][2];
#pragma unroll
  for (int g = 0; g < 2; ++g) {
    qB[g][0] = load_b_frag(Qb, HD, s0 + 16 * g, 0,  lane);
    qB[g][1] = load_b_frag(Qb, HD, s0 + 16 * g, 32, lane);
  }

  float m_run[2], l_run[2];
  v8f acc[2][4];
  v8f z = {};
#pragma unroll
  for (int g = 0; g < 2; ++g) {
    m_run[g] = -3.0e38f;
    l_run[g] = 0.0f;
#pragma unroll
    for (int j = 0; j < 4; ++j) acc[g][j] = z;
  }

  for (int tb = 0; tb < SDIM; tb += 32) {
    // K tiles as A-matrix (shared by both row groups): 2 t-subtiles x 2 d-steps.
    v16bf kA00 = load_a_frag(Kb, HD, tb,      0,  lane);
    v16bf kA01 = load_a_frag(Kb, HD, tb,      32, lane);
    v16bf kA10 = load_a_frag(Kb, HD, tb + 16, 0,  lane);
    v16bf kA11 = load_a_frag(Kb, HD, tb + 16, 32, lane);
    // V fragments (shared by both row groups): B(k=t, n=hd) from transposed V rows.
    v16bf vF[4];
#pragma unroll
    for (int j = 0; j < 4; ++j) vF[j] = load_b_frag(Vb, SDIM, j * 16, tb, lane);

#pragma unroll
    for (int g = 0; g < 2; ++g) {
      // scT0: rows t = tb..tb+15 (this lane holds t-subset per half), cols s (lane-owned).
      v8f scT0 = __builtin_amdgcn_wmma_f32_16x16x32_bf16(false, kA00, false, qB[g][0], (short)0, z,    false, false);
      scT0     = __builtin_amdgcn_wmma_f32_16x16x32_bf16(false, kA01, false, qB[g][1], (short)0, scT0, false, false);
      v8f scT1 = __builtin_amdgcn_wmma_f32_16x16x32_bf16(false, kA10, false, qB[g][0], (short)0, z,    false, false);
      scT1     = __builtin_amdgcn_wmma_f32_16x16x32_bf16(false, kA11, false, qB[g][1], (short)0, scT1, false, false);

      // Per-lane: 16 score values for query row s = s0 + 16g + (lane&15).
      float sv[16];
#pragma unroll
      for (int e = 0; e < 8; ++e) { sv[e] = scT0[e]; sv[8 + e] = scT1[e]; }

      float tmax = sv[0];
#pragma unroll
      for (int e = 1; e < 16; ++e) tmax = fmaxf(tmax, sv[e]);
      tmax = fmaxf(tmax, __shfl_xor(tmax, 16, 32));   // combine the two t-halves of the row

      float mn = fmaxf(m_run[g], tmax);
      float corr = __expf(m_run[g] - mn);
      m_run[g] = mn;

      float p[16], lsum = 0.0f;
#pragma unroll
      for (int e = 0; e < 16; ++e) { p[e] = __expf(sv[e] - mn); lsum += p[e]; }
      lsum += __shfl_xor(lsum, 16, 32);
      l_run[g] = l_run[g] * corr + lsum;

      // Pack P directly into A-fragment layout (S^T C-layout == P A-layout per lane).
      FragU pf;
#pragma unroll
      for (int i = 0; i < 8; ++i)
        pf.u[i] = (unsigned)f2bf(p[2 * i]) | ((unsigned)f2bf(p[2 * i + 1]) << 16);

      // Broadcast per-row corr from lane-indexed rows to VGPR-indexed accumulator rows.
#pragma unroll
      for (int v = 0; v < 8; ++v) {
        float cr = __shfl(corr, v + 8 * half, 32);
#pragma unroll
        for (int j = 0; j < 4; ++j) acc[g][j][v] *= cr;
      }

      // O_g += P_g[16 x 32t] * V[32t x 64hd]
#pragma unroll
      for (int j = 0; j < 4; ++j)
        acc[g][j] = __builtin_amdgcn_wmma_f32_16x16x32_bf16(false, pf.v, false, vF[j],
                                                            (short)0, acc[g][j], false, false);
    }
  }

  // Normalize and emit context (bf16, [S,B,E] layout).
#pragma unroll
  for (int g = 0; g < 2; ++g) {
    float linv = 1.0f / l_run[g];                      // lane-indexed row
#pragma unroll
    for (int v = 0; v < 8; ++v) {
      float lv = __shfl(linv, v + 8 * half, 32);       // to VGPR-indexed row
      int s = s0 + 16 * g + v + 8 * half;
      long rowoff = ((long)s * BDIM + b) * EDIM + (long)h * HD;
#pragma unroll
      for (int j = 0; j < 4; ++j)
        ctx[rowoff + j * 16 + ln] = f2bf(acc[g][j][v] * lv);
    }
  }
}

extern "C" void kernel_launch(void* const* d_in, const int* in_sizes, int n_in,
                              void* d_out, int out_size, void* d_ws, size_t ws_size,
                              hipStream_t stream) {
  const float* query = (const float*)d_in[0];
  const float* key_  = (const float*)d_in[1];
  const float* value = (const float*)d_in[2];
  const float* Wq = (const float*)d_in[3];
  const float* bq = (const float*)d_in[4];
  const float* Wk = (const float*)d_in[5];
  const float* bk = (const float*)d_in[6];
  const float* Wv = (const float*)d_in[7];
  const float* bv = (const float*)d_in[8];
  const float* Wo = (const float*)d_in[9];
  const float* bo = (const float*)d_in[10];
  float* out = (float*)d_out;

  const long NSE = (long)SDIM * BDIM * EDIM;  // 4Mi activation elems
  const long NEE = (long)EDIM * EDIM;         // 1Mi weight elems

  unsigned short* w   = (unsigned short*)d_ws;
  unsigned short* qb  = w;                    // bf16 query  [S*B, E]
  unsigned short* kbm = w + NSE;              // bf16 key
  unsigned short* vbm = w + 2 * NSE;          // bf16 value
  unsigned short* wqb = w + 3 * NSE;
  unsigned short* wkb = wqb + NEE;
  unsigned short* wvb = wkb + NEE;
  unsigned short* wob = wvb + NEE;
  unsigned short* Qp  = wob + NEE;            // [B,H,S,HD]
  unsigned short* Kp  = Qp + NSE;             // [B,H,S,HD]
  unsigned short* Vt  = Kp + NSE;             // [B,H,HD,S]
  unsigned short* ctx = Vt + NSE;             // [S,B,E]

  const int thr = 256;
  cvt_f32_bf16<<<(int)((NSE + thr - 1) / thr), thr, 0, stream>>>(query, qb, (int)NSE);
  cvt_f32_bf16<<<(int)((NSE + thr - 1) / thr), thr, 0, stream>>>(key_,  kbm, (int)NSE);
  cvt_f32_bf16<<<(int)((NSE + thr - 1) / thr), thr, 0, stream>>>(value, vbm, (int)NSE);
  cvt_f32_bf16<<<(int)((NEE + thr - 1) / thr), thr, 0, stream>>>(Wq, wqb, (int)NEE);
  cvt_f32_bf16<<<(int)((NEE + thr - 1) / thr), thr, 0, stream>>>(Wk, wkb, (int)NEE);
  cvt_f32_bf16<<<(int)((NEE + thr - 1) / thr), thr, 0, stream>>>(Wv, wvb, (int)NEE);
  cvt_f32_bf16<<<(int)((NEE + thr - 1) / thr), thr, 0, stream>>>(Wo, wob, (int)NEE);

  dim3 gg(EDIM / 64, (SDIM * BDIM) / 128);    // (16, 32): 128 thr = 4 waves, 128x64 per block
  const float scaling = 0.125f;               // HD^-0.5 = 64^-0.5
  gemm_wmma<<<gg, 128, 0, stream>>>(qb,  wqb, bq, scaling, EDIM, 1, Qp, nullptr);
  gemm_wmma<<<gg, 128, 0, stream>>>(kbm, wkb, bk, 1.0f,    EDIM, 1, Kp, nullptr);
  gemm_wmma<<<gg, 128, 0, stream>>>(vbm, wvb, bv, 1.0f,    EDIM, 2, Vt, nullptr);

  attn_wmma<<<dim3(SDIM / 128, BDIM * HDIM), 128, 0, stream>>>(Qp, Kp, Vt, ctx);

  gemm_wmma<<<gg, 128, 0, stream>>>(ctx, wob, bo, 1.0f, EDIM, 0, nullptr, out);
}
